// ASFF_2_223338299760
// MI455X (gfx1250) — matrix operand order repaired and checked by reading, not compiled, not run on hardware
//
#include <hip/hip_runtime.h>
#include <hip/hip_bf16.h>

// ---------------------------------------------------------------------------
// Types for CDNA5 WMMA (wave32): D = A(16x32 bf16) * B(32x16 bf16) + C(f32)
// ---------------------------------------------------------------------------
typedef __attribute__((ext_vector_type(16))) __bf16 v16bf;
typedef __attribute__((ext_vector_type(8)))  __bf16 v8bf;
typedef __attribute__((ext_vector_type(8)))  float  v8f;

__device__ inline v8f v8f_zero() {
    v8f r;
#pragma unroll
    for (int i = 0; i < 8; ++i) r[i] = 0.f;
    return r;
}

__device__ inline v8f wmma_bf16(v16bf a, v16bf b, v8f c) {
    // emits v_wmma_f32_16x16x32_bf16
    return __builtin_amdgcn_wmma_f32_16x16x32_bf16(false, a, false, b, (short)0, c, false, false);
}

// Async DMA: global -> LDS, 2 x 16B per call (ASYNCcnt += 2 for the wave).
// LDS destination passed as low-32 bits of the flat address of a __shared__
// object; INST_OFFSET applies to both sides.
__device__ inline void async_copy32(unsigned lds_addr, const void* gaddr) {
    asm volatile("global_load_async_to_lds_b128 %0, %1, off\n\t"
                 "global_load_async_to_lds_b128 %0, %1, off offset:16"
                 :: "v"(lds_addr), "v"((unsigned long long)(uintptr_t)gaddr)
                 : "memory");
}
__device__ inline void async_copy16(unsigned lds_addr, const void* gaddr) {
    asm volatile("global_load_async_to_lds_b128 %0, %1, off"
                 :: "v"(lds_addr), "v"((unsigned long long)(uintptr_t)gaddr)
                 : "memory");
}
// Per-stage per-wave async count is uniformly 3 (2 for A + 1 for B), so with
// ping-pong buffers: wait <=3 keeps next stage in flight, current complete.
__device__ inline void async_wait_prev() { asm volatile("s_wait_asynccnt 0x3" ::: "memory"); }
__device__ inline void async_wait_all()  { asm volatile("s_wait_asynccnt 0x0" ::: "memory"); }

// LDS row stride (bf16 elements) for A/B staging: 32 data + 8 pad, keeps 16B alignment
#define LDS_STRIDE 40

// A fragment: 16x32 bf16. lane holds row m = base + (lane&15);
// K elements: i<8 -> kb+i ; i>=8 -> kb+16+(i-8), kb = (lane>>4)*8.
__device__ inline v16bf load_a_frag(const __bf16* lds, int row_base, int lane) {
    int m  = row_base + (lane & 15);
    int kb = (lane >> 4) << 3;
    const __bf16* p = lds + m * LDS_STRIDE + kb;
    v8bf lo = *reinterpret_cast<const v8bf*>(p);
    v8bf hi = *reinterpret_cast<const v8bf*>(p + 16);
    v16bf f;
#pragma unroll
    for (int i = 0; i < 8; ++i) { f[i] = lo[i]; f[i + 8] = hi[i]; }
    return f;
}

// B fragment: 32x16 bf16. lane holds col n = base + (lane&15);
// K elements: i<8 -> kb+i ; i>=8 -> kb+16+(i-8), kb = (lane>>4)*8. Stored n-major (B^T) in LDS.
__device__ inline v16bf load_b_frag(const __bf16* lds, int row_base, int lane) {
    int n  = row_base + (lane & 15);
    int kb = (lane >> 4) << 3;
    const __bf16* p = lds + n * LDS_STRIDE + kb;
    v8bf lo = *reinterpret_cast<const v8bf*>(p);
    v8bf hi = *reinterpret_cast<const v8bf*>(p + 16);
    v16bf f;
#pragma unroll
    for (int i = 0; i < 8; ++i) { f[i] = lo[i]; f[i + 8] = hi[i]; }
    return f;
}

// ---------------------------------------------------------------------------
// Prep kernels: NCHW->NHWC transposes (LDS tiled) and weight bf16 converts
// ---------------------------------------------------------------------------
__global__ __launch_bounds__(256) void tr_tile_f32_kernel(const float* __restrict__ in,
                                                          float* __restrict__ out,
                                                          int C, int P) {
    __shared__ float t[32][33];
    int n = blockIdx.z, cb = blockIdx.y * 32, pb = blockIdx.x * 32;
    int tx = threadIdx.x & 31, ty = threadIdx.x >> 5;  // 32 x 8
#pragma unroll
    for (int k = 0; k < 32; k += 8)
        t[ty + k][tx] = in[((size_t)n * C + cb + ty + k) * P + pb + tx];
    __syncthreads();
#pragma unroll
    for (int k = 0; k < 32; k += 8)
        out[((size_t)n * P + pb + ty + k) * C + cb + tx] = t[tx][ty + k];
}

__global__ __launch_bounds__(256) void tr_tile_bf16_kernel(const float* __restrict__ in,
                                                           __bf16* __restrict__ out,
                                                           int C, int P) {
    __shared__ float t[32][33];
    int n = blockIdx.z, cb = blockIdx.y * 32, pb = blockIdx.x * 32;
    int tx = threadIdx.x & 31, ty = threadIdx.x >> 5;
#pragma unroll
    for (int k = 0; k < 32; k += 8)
        t[ty + k][tx] = in[((size_t)n * C + cb + ty + k) * P + pb + tx];
    __syncthreads();
#pragma unroll
    for (int k = 0; k < 32; k += 8)
        out[((size_t)n * P + pb + ty + k) * C + cb + tx] = (__bf16)t[tx][ty + k];
}

__global__ __launch_bounds__(256) void cvt_f32_bf16_kernel(const float* __restrict__ in,
                                                           __bf16* __restrict__ out, int n) {
    int i = blockIdx.x * 256 + threadIdx.x;
    if (i < n) out[i] = (__bf16)in[i];
}

// w_conv (O=256, C=256, 3, 3) -> wq[tap][o][c] bf16
__global__ __launch_bounds__(256) void wconv_prep_kernel(const float* __restrict__ w,
                                                         __bf16* __restrict__ out) {
    int i = blockIdx.x * 256 + threadIdx.x;  // 589824
    int c = i & 255, o = (i >> 8) & 255, t = i >> 16;
    out[i] = (__bf16)w[(size_t)(((o << 8) | c) * 9) + t];
}

// Zero the 1-pixel halo of the padded fused buffer [8][66][66][256] bf16
__global__ __launch_bounds__(256) void halo_zero_kernel(__bf16* __restrict__ fusedp) {
    int idx = blockIdx.x * 256 + threadIdx.x;  // pixels: 8*66*66 = 34848
    if (idx >= 34848) return;
    int r = idx % 4356;
    int y = r / 66, x = r % 66;
    if (y == 0 || y == 65 || x == 0 || x == 65) {
        v8bf z;
#pragma unroll
        for (int j = 0; j < 8; ++j) z[j] = (__bf16)0.f;
        __bf16* p = fusedp + (size_t)idx * 256;
        for (int c = 0; c < 256; c += 8)
            *reinterpret_cast<v8bf*>(p + c) = z;
    }
}

// ---------------------------------------------------------------------------
// Generic bf16 WMMA GEMM: Out[m][o] = sum_k A[m][k]*B[o][k] + bias[o]
// Block: 256 thr (8 waves, 4x2), tile 128M x 64O, K step 32.
// Double-buffered LDS fed by async global->LDS DMA (ASYNCcnt ping-pong).
// ---------------------------------------------------------------------------
__global__ __launch_bounds__(256) void gemm_bf16_kernel(const __bf16* __restrict__ A,
                                                        const __bf16* __restrict__ B,
                                                        const float* __restrict__ bias,
                                                        float* __restrict__ Out,
                                                        int K, int Ocols) {
    __shared__ __bf16 lds_a[2][128 * LDS_STRIDE];
    __shared__ __bf16 lds_b[2][64 * LDS_STRIDE];
    const int tid = threadIdx.x, lane = tid & 31, wv = tid >> 5;
    const int wm = wv & 3, wn = wv >> 2;
    const int mbase = blockIdx.x * 128;
    const int obase = blockIdx.y * 64;

    // per-thread staging slots
    const int arow = tid >> 1, ahalf = tid & 1;  // A: 128 rows x 64B, 32B/thread
    const int brow = tid >> 2, bq = tid & 3;     // B: 64 rows x 64B, 16B/thread
    const __bf16* gA = A + (size_t)(mbase + arow) * K + ahalf * 16;
    const __bf16* gB = B + (size_t)(obase + brow) * K + bq * 8;
    unsigned la[2], lb[2];
#pragma unroll
    for (int b = 0; b < 2; ++b) {
        la[b] = (unsigned)(uintptr_t)&lds_a[b][arow * LDS_STRIDE + ahalf * 16];
        lb[b] = (unsigned)(uintptr_t)&lds_b[b][brow * LDS_STRIDE + bq * 8];
    }

    v8f acc[2][2];
#pragma unroll
    for (int i = 0; i < 2; ++i)
#pragma unroll
        for (int j = 0; j < 2; ++j) acc[i][j] = v8f_zero();

    auto issue = [&](int k0, int buf) {
        async_copy32(la[buf], gA + k0);
        async_copy16(lb[buf], gB + k0);
    };

    const int nit = K >> 5;
    issue(0, 0);
    for (int it = 0; it < nit; ++it) {
        const int cur = it & 1;
        if (it + 1 < nit) {
            issue((it + 1) << 5, cur ^ 1);
            async_wait_prev();   // next stage (3 asyncs) in flight, current done
        } else {
            async_wait_all();
        }
        __syncthreads();
        v16bf a0 = load_a_frag(lds_a[cur], wm * 32, lane);
        v16bf a1 = load_a_frag(lds_a[cur], wm * 32 + 16, lane);
        v16bf b0 = load_b_frag(lds_b[cur], wn * 32, lane);
        v16bf b1 = load_b_frag(lds_b[cur], wn * 32 + 16, lane);
        acc[0][0] = wmma_bf16(a0, b0, acc[0][0]);
        acc[0][1] = wmma_bf16(a0, b1, acc[0][1]);
        acc[1][0] = wmma_bf16(a1, b0, acc[1][0]);
        acc[1][1] = wmma_bf16(a1, b1, acc[1][1]);
        __syncthreads();
    }

    // Epilogue: C/D layout -> lanes 0-15: M=r, lanes 16-31: M=r+8, N=lane&15
#pragma unroll
    for (int i = 0; i < 2; ++i)
#pragma unroll
        for (int j = 0; j < 2; ++j) {
            int o = obase + wn * 32 + j * 16 + (lane & 15);
            float bo = bias[o];
            int mrow = mbase + wm * 32 + i * 16 + ((lane >> 4) << 3);
#pragma unroll
            for (int r = 0; r < 8; ++r)
                Out[(size_t)(mrow + r) * Ocols + o] = acc[i][j][r] + bo;
        }
}

// ---------------------------------------------------------------------------
// enc conv 3x3 (128->36) + pixel-shuffle regroup + softmax over the 9 taps.
// Thread per (n,q,h,w); writes mask[n][q][k][32][32].
// ---------------------------------------------------------------------------
__global__ __launch_bounds__(256) void enc_softmax_kernel(const float* __restrict__ kt0,   // NHWC [8192][128]
                                                          const float* __restrict__ w_enc, // (36,128,3,3)
                                                          const float* __restrict__ b_enc,
                                                          float* __restrict__ mask) {
    int idx = blockIdx.x * 256 + threadIdx.x;  // 32768
    int w = idx & 31, h = (idx >> 5) & 31, q = (idx >> 10) & 3, n = idx >> 12;
    float acc[9];
#pragma unroll
    for (int k = 0; k < 9; ++k) acc[k] = b_enc[k * 4 + q];
    for (int dh = 0; dh < 3; ++dh) {
        int hh = h + dh - 1;
        if ((unsigned)hh >= 32u) continue;
        for (int dw = 0; dw < 3; ++dw) {
            int ww = w + dw - 1;
            if ((unsigned)ww >= 32u) continue;
            const float* xp = kt0 + (size_t)((n * 32 + hh) * 32 + ww) * 128;
            for (int c = 0; c < 128; ++c) {
                float x = xp[c];
#pragma unroll
                for (int k = 0; k < 9; ++k)
                    acc[k] += x * w_enc[(size_t)(((k * 4 + q) * 128 + c) * 3 + dh) * 3 + dw];
            }
        }
    }
    float mx = acc[0];
#pragma unroll
    for (int k = 1; k < 9; ++k) mx = fmaxf(mx, acc[k]);
    float s = 0.f;
#pragma unroll
    for (int k = 0; k < 9; ++k) { acc[k] = __expf(acc[k] - mx); s += acc[k]; }
    float inv = 1.f / s;
#pragma unroll
    for (int k = 0; k < 9; ++k)
        mask[(((size_t)(n * 4 + q) * 9 + k) << 10) + (h << 5) + w] = acc[k] * inv;
}

// ---------------------------------------------------------------------------
// CARAFE reassembly: mid2[n][2h+a][2w+b][c] = sum_k in2[n][h+dh][w+dw][c]*m[k][q]
// ---------------------------------------------------------------------------
__global__ __launch_bounds__(256) void carafe_kernel(const __bf16* __restrict__ in2t,   // NHWC [8192][512]
                                                     const float* __restrict__ mask,
                                                     __bf16* __restrict__ mid2) {      // NHWC [32768][512]
    int idx = blockIdx.x * 256 + threadIdx.x;  // 524288
    int cc = idx & 63;
    int pix = idx >> 6;
    int w = pix & 31, h = (pix >> 5) & 31, n = pix >> 10;

    float m[9][4];
#pragma unroll
    for (int q = 0; q < 4; ++q)
#pragma unroll
        for (int k = 0; k < 9; ++k)
            m[k][q] = mask[(((size_t)(n * 4 + q) * 9 + k) << 10) + (h << 5) + w];

    float o[8][4];
#pragma unroll
    for (int j = 0; j < 8; ++j)
#pragma unroll
        for (int q = 0; q < 4; ++q) o[j][q] = 0.f;

#pragma unroll
    for (int k = 0; k < 9; ++k) {
        int hh = h + k / 3 - 1, ww = w + k % 3 - 1;
        if ((unsigned)hh >= 32u || (unsigned)ww >= 32u) continue;
        v8bf p = *reinterpret_cast<const v8bf*>(in2t + (size_t)((n * 32 + hh) * 32 + ww) * 512 + cc * 8);
#pragma unroll
        for (int j = 0; j < 8; ++j) {
            float x = (float)p[j];
#pragma unroll
            for (int q = 0; q < 4; ++q) o[j][q] += x * m[k][q];
        }
    }
#pragma unroll
    for (int q = 0; q < 4; ++q) {
        int y = 2 * h + (q >> 1), x2 = 2 * w + (q & 1);
        v8bf pv;
#pragma unroll
        for (int j = 0; j < 8; ++j) pv[j] = (__bf16)o[j][q];
        *reinterpret_cast<v8bf*>(mid2 + (size_t)((n * 64 + y) * 64 + x2) * 512 + cc * 8) = pv;
    }
}

// ---------------------------------------------------------------------------
// Level weights: v1=BNSiLU(1x1(in1)), v2=BNSiLU(1x1(in2u)), lw=softmax2(Wl[v1;v2]+b)
// ---------------------------------------------------------------------------
__global__ __launch_bounds__(256) void lw_kernel(const float* __restrict__ in1t,  // NHWC [32768][256]
                                                 const float* __restrict__ in2u,  // NHWC [32768][256]
                                                 const float* __restrict__ w_l1,
                                                 const float* __restrict__ bn_l1,
                                                 const float* __restrict__ w_l2,
                                                 const float* __restrict__ bn_l2,
                                                 const float* __restrict__ w_wl,
                                                 const float* __restrict__ b_wl,
                                                 float* __restrict__ lw) {
    __shared__ float sw1[2048], sw2[2048];
    for (int i = threadIdx.x; i < 2048; i += 256) { sw1[i] = w_l1[i]; sw2[i] = w_l2[i]; }
    __syncthreads();
    int pix = blockIdx.x * 256 + threadIdx.x;  // 32768
    const float* x1 = in1t + (size_t)pix * 256;
    const float* x2 = in2u + (size_t)pix * 256;
    float a1[8], a2[8];
#pragma unroll
    for (int o = 0; o < 8; ++o) { a1[o] = 0.f; a2[o] = 0.f; }
    for (int c = 0; c < 256; ++c) {
        float u = x1[c], v = x2[c];
#pragma unroll
        for (int o = 0; o < 8; ++o) {
            a1[o] += u * sw1[o * 256 + c];
            a2[o] += v * sw2[o * 256 + c];
        }
    }
    float vc[16];
#pragma unroll
    for (int o = 0; o < 8; ++o) {
        float y = (a1[o] - bn_l1[16 + o]) * (bn_l1[o] * rsqrtf(bn_l1[24 + o] + 1e-5f)) + bn_l1[8 + o];
        vc[o] = y / (1.f + __expf(-y));
        float z = (a2[o] - bn_l2[16 + o]) * (bn_l2[o] * rsqrtf(bn_l2[24 + o] + 1e-5f)) + bn_l2[8 + o];
        vc[8 + o] = z / (1.f + __expf(-z));
    }
    float l0 = b_wl[0], l1 = b_wl[1];
#pragma unroll
    for (int i = 0; i < 16; ++i) { l0 += vc[i] * w_wl[i]; l1 += vc[i] * w_wl[16 + i]; }
    float mx = fmaxf(l0, l1);
    float e0 = __expf(l0 - mx), e1 = __expf(l1 - mx);
    float inv = 1.f / (e0 + e1);
    lw[pix * 2] = e0 * inv;
    lw[pix * 2 + 1] = e1 * inv;
}

// fused = in1*l0 + in2u*l1, written into the zero-padded [8][66][66][256] bf16
// buffer (interior only). Block = one pixel's 256 channels.
__global__ __launch_bounds__(256) void fuse_kernel(const float* __restrict__ in1t,
                                                   const float* __restrict__ in2u,
                                                   const float* __restrict__ lw,
                                                   __bf16* __restrict__ fusedp) {
    int pix = blockIdx.x, c = threadIdx.x;
    int n = pix >> 12, hw = pix & 4095, h = hw >> 6, w = hw & 63;
    float l0 = lw[pix * 2], l1 = lw[pix * 2 + 1];
    size_t i = (size_t)pix * 256 + c;
    fusedp[((size_t)((n * 66 + h + 1) * 66) + w + 1) * 256 + c] =
        (__bf16)(in1t[i] * l0 + in2u[i] * l1);
}

// ---------------------------------------------------------------------------
// Final 3x3 conv 256->256 pad=1 as implicit GEMM + BN + SiLU -> NCHW fp32 out.
// Halo-padded input makes every A-load in-bounds -> branch-free inner loop
// with uniform async counts (2 A + 1 B per wave per stage), double-buffered.
// Epilogue transposes each 16x16 tile through LDS for coalesced NCHW stores.
// ---------------------------------------------------------------------------
__global__ __launch_bounds__(256) void conv3x3_wmma_kernel(const __bf16* __restrict__ fusedp, // [8][66][66][256]
                                                           const __bf16* __restrict__ wq,     // [9][256][256]
                                                           const float* __restrict__ bn,      // [4][256]
                                                           float* __restrict__ out) {         // NCHW [8][256][64][64]
    __shared__ __bf16 lds_a[2][128 * LDS_STRIDE];
    __shared__ __bf16 lds_b[2][64 * LDS_STRIDE];
    __shared__ float  lds_t[8 * 272];  // per-wave 16x17 transpose buffer

    const int tid = threadIdx.x, lane = tid & 31, wv = tid >> 5;
    const int wm = wv & 3, wn = wv >> 2;
    const int nimg = blockIdx.x >> 5;
    const int h0 = (blockIdx.x & 31) * 2;
    const int obase = blockIdx.y * 64;

    const int arow = tid >> 1, ahalf = tid & 1;
    const int brow = tid >> 2, bq = tid & 3;
    unsigned la[2], lb[2];
#pragma unroll
    for (int b = 0; b < 2; ++b) {
        la[b] = (unsigned)(uintptr_t)&lds_a[b][arow * LDS_STRIDE + ahalf * 16];
        lb[b] = (unsigned)(uintptr_t)&lds_b[b][brow * LDS_STRIDE + bq * 8];
    }

    v8f acc[2][2];
#pragma unroll
    for (int i = 0; i < 2; ++i)
#pragma unroll
        for (int j = 0; j < 2; ++j) acc[i][j] = v8f_zero();

    // stage in [0,72): tap = stage/8 (3x3 position), c0 = (stage%8)*32
    auto issue = [&](int stage, int buf) {
        int tap = stage >> 3, c0 = (stage & 7) << 5;
        int dh = tap / 3, dw = tap % 3;
        int hh = h0 + (arow >> 6) + dh;  // padded coords, always in [0,65]
        int ww = (arow & 63) + dw;
        const __bf16* gA = fusedp + (size_t)((nimg * 66 + hh) * 66 + ww) * 256 + c0 + ahalf * 16;
        async_copy32(la[buf], gA);
        const __bf16* gB = wq + ((size_t)tap * 256 + obase + brow) * 256 + c0 + bq * 8;
        async_copy16(lb[buf], gB);
    };

    issue(0, 0);
    for (int it = 0; it < 72; ++it) {
        const int cur = it & 1;
        if (it + 1 < 72) {
            issue(it + 1, cur ^ 1);
            async_wait_prev();
        } else {
            async_wait_all();
        }
        __syncthreads();
        v16bf a0 = load_a_frag(lds_a[cur], wm * 32, lane);
        v16bf a1 = load_a_frag(lds_a[cur], wm * 32 + 16, lane);
        v16bf b0 = load_b_frag(lds_b[cur], wn * 32, lane);
        v16bf b1 = load_b_frag(lds_b[cur], wn * 32 + 16, lane);
        acc[0][0] = wmma_bf16(a0, b0, acc[0][0]);
        acc[0][1] = wmma_bf16(a0, b1, acc[0][1]);
        acc[1][0] = wmma_bf16(a1, b0, acc[1][0]);
        acc[1][1] = wmma_bf16(a1, b1, acc[1][1]);
        __syncthreads();
    }

    // Epilogue: transpose 16x16 tiles via LDS, fuse BN+SiLU, coalesced NCHW stores
#pragma unroll
    for (int i = 0; i < 2; ++i)
#pragma unroll
        for (int j = 0; j < 2; ++j) {
            {
                int ml = (lane >> 4) << 3, nl = lane & 15;
#pragma unroll
                for (int r = 0; r < 8; ++r)
                    lds_t[wv * 272 + (ml + r) * 17 + nl] = acc[i][j][r];
            }
            __syncthreads();
            {
                int ml = lane & 15;
                int row128 = wm * 32 + i * 16 + ml;
                int hh = h0 + (row128 >> 6);
                int ww = row128 & 63;
#pragma unroll
                for (int jj = 0; jj < 8; ++jj) {
                    int oc = jj * 2 + (lane >> 4);
                    float v = lds_t[wv * 272 + ml * 17 + oc];
                    int o = obase + wn * 32 + j * 16 + oc;
                    float y = (v - bn[512 + o]) * (bn[o] * rsqrtf(bn[768 + o] + 1e-5f)) + bn[256 + o];
                    out[((size_t)(nimg * 256 + o) << 12) + hh * 64 + ww] = y / (1.f + __expf(-y));
                }
            }
            __syncthreads();
        }
}

// ---------------------------------------------------------------------------
// Workspace layout (bytes, all 256-aligned):
// ---------------------------------------------------------------------------
#define WS_IN1T   0u          // f32  8*4096*256          = 33554432 B
#define WS_IN2T   33554432u   // bf16 8*1024*512          =  8388608 B
#define WS_WQD    41943040u   // bf16 128*512             =   131072 B
#define WS_WQU    42074112u   // bf16 256*512             =   262144 B
#define WS_WQC    42336256u   // bf16 9*256*256           =  1179648 B
#define WS_KT0    43515904u   // f32  8192*128            =  4194304 B
#define WS_MASK   47710208u   // f32  8*4*9*1024          =  1179648 B
#define WS_MID2   48889856u   // bf16 32768*512           = 33554432 B
#define WS_IN2U   82444288u   // f32  32768*256           = 33554432 B
#define WS_LW     115998720u  // f32  32768*2             =   262144 B
#define WS_FUSED  116260864u  // bf16 8*66*66*256         = 17842176 B
// total = 134103040 B (~128 MB)

extern "C" void kernel_launch(void* const* d_in, const int* in_sizes, int n_in,
                              void* d_out, int out_size, void* d_ws, size_t ws_size,
                              hipStream_t stream) {
    (void)in_sizes; (void)n_in; (void)out_size; (void)ws_size;
    const float* input1  = (const float*)d_in[0];
    const float* input2  = (const float*)d_in[1];
    const float* w_down  = (const float*)d_in[2];
    const float* b_down  = (const float*)d_in[3];
    const float* w_enc   = (const float*)d_in[4];
    const float* b_enc   = (const float*)d_in[5];
    const float* w_up    = (const float*)d_in[6];
    const float* b_up    = (const float*)d_in[7];
    const float* w_l1    = (const float*)d_in[8];
    const float* bn_l1   = (const float*)d_in[9];
    const float* w_l2    = (const float*)d_in[10];
    const float* bn_l2   = (const float*)d_in[11];
    const float* w_wl    = (const float*)d_in[12];
    const float* b_wl    = (const float*)d_in[13];
    const float* w_conv  = (const float*)d_in[14];
    const float* bn_conv = (const float*)d_in[15];
    float* out = (float*)d_out;

    char* ws = (char*)d_ws;
    float*  in1t  = (float*)(ws + WS_IN1T);
    __bf16* in2t  = (__bf16*)(ws + WS_IN2T);
    __bf16* wqd   = (__bf16*)(ws + WS_WQD);
    __bf16* wqu   = (__bf16*)(ws + WS_WQU);
    __bf16* wqc   = (__bf16*)(ws + WS_WQC);
    float*  kt0   = (float*)(ws + WS_KT0);
    float*  maskp = (float*)(ws + WS_MASK);
    __bf16* mid2  = (__bf16*)(ws + WS_MID2);
    float*  in2u  = (float*)(ws + WS_IN2U);
    float*  lwp   = (float*)(ws + WS_LW);
    __bf16* fusedp = (__bf16*)(ws + WS_FUSED);

    // 1) layout prep
    tr_tile_f32_kernel<<<dim3(128, 8, 8), 256, 0, stream>>>(input1, in1t, 256, 4096);
    tr_tile_bf16_kernel<<<dim3(32, 16, 8), 256, 0, stream>>>(input2, in2t, 512, 1024);
    cvt_f32_bf16_kernel<<<256, 256, 0, stream>>>(w_down, wqd, 65536);
    cvt_f32_bf16_kernel<<<512, 256, 0, stream>>>(w_up, wqu, 131072);
    wconv_prep_kernel<<<2304, 256, 0, stream>>>(w_conv, wqc);
    halo_zero_kernel<<<137, 256, 0, stream>>>(fusedp);

    // 2) CARAFE kernel-prediction branch
    gemm_bf16_kernel<<<dim3(64, 2), 256, 0, stream>>>(in2t, wqd, b_down, kt0, 512, 128);
    enc_softmax_kernel<<<128, 256, 0, stream>>>(kt0, w_enc, b_enc, maskp);
    carafe_kernel<<<2048, 256, 0, stream>>>(in2t, maskp, mid2);

    // 3) 1x1 up_out conv (512->256 @ 64x64)
    gemm_bf16_kernel<<<dim3(256, 4), 256, 0, stream>>>(mid2, wqu, b_up, in2u, 512, 256);

    // 4) level-weight softmax + fusion
    lw_kernel<<<128, 256, 0, stream>>>(in1t, in2u, w_l1, bn_l1, w_l2, bn_l2, w_wl, b_wl, lwp);
    fuse_kernel<<<32768, 256, 0, stream>>>(in1t, in2u, lwp, fusedp);

    // 5) final 3x3 conv + BN + SiLU -> d_out (NCHW f32)
    conv3x3_wmma_kernel<<<dim3(256, 4), 256, 0, stream>>>(fusedp, wqc, bn_conv, out);
}
